// Encoder_10986526343892
// MI455X (gfx1250) — matrix operand order repaired and checked by reading, not compiled, not run on hardware
//
#include <hip/hip_runtime.h>
#include <hip/hip_bf16.h>

typedef __attribute__((ext_vector_type(16))) _Float16 v16h;
typedef __attribute__((ext_vector_type(8)))  float    v8f;

// ---------------------------------------------------------------------------
// Implicit-GEMM convolution on v_wmma_f32_16x16x32_f16 (wave32).
//   M = Cout, N = output pixels, K = Cin*KK*KK (zero-padded to multiple of 32)
//
// Block = 256 threads = 8 waves arranged 2 (M) x 4 (N); each wave computes a
// 2x2 grid of 16x16 WMMA tiles -> block tile = 64 Cout x 128 pixels,
// 32 v_wmma issues per staged K-chunk. LDS staging is double-buffered: while
// chunk k is consumed by the matrix pipe, chunk k+1's global loads issue.
//
// Activations are f16 in global memory (produced by inorm_act), weights f32
// (converted at stage time; tiny and L2-resident). Accumulation f32.
//
// LDS tiles are stored in WMMA *fragment order* so each lane's operand load
// is one contiguous 32B read:
//   A (16x32 f16): lane l, elem j -> row m = l&15,
//                  k = (j<8 ? j + 8*(l>>4) : j + 8 + 8*(l>>4))
//   B (32x16 f16): lane l, elem j -> k = (j<8 ? j + 8*(l>>4) : j + 8 + 8*(l>>4)), col n = l&15
//   C/D (16x16 f32): reg r, lane l -> m = r + 8*(l>>4), n = l&15
//
// TR==1 implements ConvTranspose2d(k=3,s=2,p=1,output_padding=1) as a
// cross-correlation over the zero-dilated input (invalid parity taps read 0).
// Normal convs use reflect padding (jnp.pad mode='reflect').
// ---------------------------------------------------------------------------
template<int KK, int STRIDE, int PAD, int TR>
__global__ __launch_bounds__(256)
void conv_wmma(const _Float16* __restrict__ x, const float* __restrict__ w,
               const float* __restrict__ bias, float* __restrict__ y,
               int Cin, int Cout, int Hin, int Win, int Hout, int Wout, int wshift)
{
    constexpr int KHW = KK * KK;

    const int tid  = threadIdx.x;
    const int lane = tid & 31;
    const int wv   = tid >> 5;
    const int wm   = wv >> 2;     // 0..1  (M wave)
    const int wn   = wv & 3;      // 0..3  (N wave)

    const int pix0 = blockIdx.x * 128;
    const int co0  = blockIdx.y * 64;
    const int nb   = blockIdx.z;

    const int HWin  = Hin * Win;
    const int HWout = Hout * Wout;
    const int Ktot  = Cin * KHW;

    const _Float16* __restrict__ xb = x + (size_t)nb * Cin * HWin;
    float* __restrict__          yb = y + (size_t)nb * Cout * HWout;

    __shared__ alignas(32) _Float16 sA[2][4 * 32 * 16];   // double-buffered A
    __shared__ alignas(32) _Float16 sB[2][8 * 32 * 16];   // double-buffered B
    __shared__ int soy[128];
    __shared__ int sox[128];

    // Output-pixel coords: all spatial sizes are powers of two -> shift/mask.
    if (tid < 128) {
        const int p = pix0 + tid;
        soy[tid] = p >> wshift;
        sox[tid] = p & (Wout - 1);
    }
    __syncthreads();

    auto stageB = [&](int kc, _Float16* __restrict__ dB) {
        for (int e = tid; e < 8 * 32 * 16; e += 256) {
            const int nt   = e >> 9;          // N sub-tile 0..7
            const int rem  = e & 511;
            const int ln   = rem >> 4;        // target lane
            const int j    = rem & 15;        // target element
            const int n    = ln & 15;
            const int k    = (j < 8) ? (j + 8 * (ln >> 4)) : (j + 8 + 8 * (ln >> 4));
            const int kidx = kc + k;
            _Float16 v = (_Float16)0.0f;
            if (kidx < Ktot) {
                const int ci = kidx / KHW;             // constant divisor -> mul/shift
                const int t  = kidx - ci * KHW;
                const int ky = t / KK;
                const int kx = t - ky * KK;
                const int np = nt * 16 + n;
                const int oy = soy[np];
                const int ox = sox[np];
                if (!TR) {
                    int iy = oy * STRIDE - PAD + ky;
                    int ix = ox * STRIDE - PAD + kx;
                    iy = iy < 0 ? -iy : (iy >= Hin ? 2 * Hin - 2 - iy : iy);
                    ix = ix < 0 ? -ix : (ix >= Win ? 2 * Win - 2 - ix : ix);
                    v = xb[(size_t)ci * HWin + iy * Win + ix];
                } else {
                    const int dy = oy - 1 + ky;   // dilated coords (lhs_dilation=2, pad 1)
                    const int dx = ox - 1 + kx;
                    if (((dy | dx) & 1) == 0 && dy >= 0 && dx >= 0 &&
                        dy <= 2 * Hin - 2 && dx <= 2 * Win - 2)
                        v = xb[(size_t)ci * HWin + (dy >> 1) * Win + (dx >> 1)];
                }
            }
            dB[e] = v;
        }
    };
    auto stageA = [&](int kc, _Float16* __restrict__ dA) {
        for (int e = tid; e < 4 * 32 * 16; e += 256) {
            const int mt   = e >> 9;          // M sub-tile 0..3
            const int rem  = e & 511;
            const int ln   = rem >> 4;
            const int j    = rem & 15;
            const int m    = ln & 15;
            const int half = ln >> 4;
            const int k    = (j < 8) ? (j + 8 * half) : (j + 8 + 8 * half);
            const int kidx = kc + k;
            const int co   = co0 + mt * 16 + m;
            float v = 0.0f;
            if (co < Cout && kidx < Ktot)
                v = w[(size_t)co * Ktot + kidx];   // OIHW == [co][ci*KHW + ky*KK + kx]
            dA[e] = (_Float16)v;
        }
    };

    stageB(0, sB[0]);
    stageA(0, sA[0]);
    __syncthreads();

    v8f acc[2][2] = {{{}, {}}, {{}, {}}};
    int buf = 0;

    for (int kc = 0; kc < Ktot; kc += 32) {
        // Prefetch next chunk into the other LDS buffer (overlaps WMMA below).
        if (kc + 32 < Ktot) {
            stageB(kc + 32, sB[buf ^ 1]);
            stageA(kc + 32, sA[buf ^ 1]);
        }

        const v16h a0 = *(const v16h*)&sA[buf][((wm * 2 + 0) * 32 + lane) * 16];
        const v16h a1 = *(const v16h*)&sA[buf][((wm * 2 + 1) * 32 + lane) * 16];
        const v16h b0 = *(const v16h*)&sB[buf][((wn * 2 + 0) * 32 + lane) * 16];
        const v16h b1 = *(const v16h*)&sB[buf][((wn * 2 + 1) * 32 + lane) * 16];

        acc[0][0] = __builtin_amdgcn_wmma_f32_16x16x32_f16(false, a0, false, b0,
                        (short)0, acc[0][0], false, false);
        acc[0][1] = __builtin_amdgcn_wmma_f32_16x16x32_f16(false, a0, false, b1,
                        (short)0, acc[0][1], false, false);
        acc[1][0] = __builtin_amdgcn_wmma_f32_16x16x32_f16(false, a1, false, b0,
                        (short)0, acc[1][0], false, false);
        acc[1][1] = __builtin_amdgcn_wmma_f32_16x16x32_f16(false, a1, false, b1,
                        (short)0, acc[1][1], false, false);

        __syncthreads();   // fill(buf^1) complete; reads of buf done before refill
        buf ^= 1;
    }

    // ---- epilogue: D fragments -> NCHW fp32 (+bias; cancelled by IN anyway) ----
    const int n    = lane & 15;
    const int half = lane >> 4;
    #pragma unroll
    for (int mt = 0; mt < 2; ++mt) {
        #pragma unroll
        for (int nt = 0; nt < 2; ++nt) {
            const int p = pix0 + (wn * 2 + nt) * 16 + n;
            #pragma unroll
            for (int r = 0; r < 8; ++r) {
                const int co = co0 + (wm * 2 + mt) * 16 + r + 8 * half;
                if (co < Cout && p < HWout)
                    yb[(size_t)co * HWout + p] = acc[mt][nt][r] + bias[co];
            }
        }
    }
}

// ---------------------------------------------------------------------------
// InstanceNorm (affine=False) + activation. One block per (n,c) plane.
// act==0: ReLU, emit f16 activations (consumed by the next WMMA conv).
// act==1: tanh, emit f32 in place (consumed by the segment-mean stage).
// ---------------------------------------------------------------------------
__global__ __launch_bounds__(256)
void inorm_act(const float* __restrict__ in, float* __restrict__ out32,
               _Float16* __restrict__ out16, int HW, int act)
{
    const size_t off = (size_t)blockIdx.x * HW;
    const float* __restrict__ p = in + off;
    __shared__ float ssum[256];
    __shared__ float ssq[256];
    float s = 0.0f, q = 0.0f;
    for (int i = threadIdx.x; i < HW; i += 256) {
        const float v = p[i];
        s += v; q += v * v;
    }
    ssum[threadIdx.x] = s; ssq[threadIdx.x] = q;
    __syncthreads();
    for (int st = 128; st > 0; st >>= 1) {
        if (threadIdx.x < st) {
            ssum[threadIdx.x] += ssum[threadIdx.x + st];
            ssq [threadIdx.x] += ssq [threadIdx.x + st];
        }
        __syncthreads();
    }
    const float mean = ssum[0] / (float)HW;
    const float var  = fmaxf(ssq[0] / (float)HW - mean * mean, 0.0f);
    const float inv  = rsqrtf(var + 1e-5f);
    if (act == 0) {
        _Float16* __restrict__ o = out16 + off;
        for (int i = threadIdx.x; i < HW; i += 256)
            o[i] = (_Float16)fmaxf((p[i] - mean) * inv, 0.0f);
    } else {
        float* __restrict__ o = out32 + off;
        for (int i = threadIdx.x; i < HW; i += 256)
            o[i] = tanhf((p[i] - mean) * inv);
    }
}

// ---------------------------------------------------------------------------
// Input conversion: f32 -> f16 (first conv consumes f16 activations).
// ---------------------------------------------------------------------------
__global__ __launch_bounds__(256)
void f32_to_f16(const float* __restrict__ in, _Float16* __restrict__ out, int n)
{
    const int i = blockIdx.x * 256 + threadIdx.x;
    if (i < n) out[i] = (_Float16)in[i];
}

// ---------------------------------------------------------------------------
// Segment mean over instance ids (C=3, HW=512*512 compile-time).
// Per (n,c) block: LDS float histogram via ds_add_f32 atomics -> means table.
// ---------------------------------------------------------------------------
__global__ __launch_bounds__(256)
void imean_stats(const float* __restrict__ feat, const int* __restrict__ ids,
                 float* __restrict__ means)
{
    constexpr int C = 3;
    constexpr int HW = 1 << 18;
    const int n = blockIdx.x / C;
    const int c = blockIdx.x - n * C;
    __shared__ float ssum[256];
    __shared__ float scnt[256];
    ssum[threadIdx.x] = 0.0f;
    scnt[threadIdx.x] = 0.0f;
    __syncthreads();
    const float* __restrict__ f  = feat + ((size_t)n * C + c) * HW;
    const int*   __restrict__ id = ids + (size_t)n * HW;
    for (int i = threadIdx.x; i < HW; i += 256) {
        const int g = id[i] & 255;
        atomicAdd(&ssum[g], f[i]);
        atomicAdd(&scnt[g], 1.0f);
    }
    __syncthreads();
    means[((size_t)n * C + c) * 256 + threadIdx.x] =
        ssum[threadIdx.x] / fmaxf(scnt[threadIdx.x], 1.0f);
}

__global__ __launch_bounds__(256)
void imean_scatter(const float* __restrict__ means, const int* __restrict__ ids,
                   float* __restrict__ out)
{
    constexpr int C = 3;
    constexpr int HW = 1 << 18;
    const size_t i = (size_t)blockIdx.x * 256 + threadIdx.x;   // over N*C*HW
    const int n = (int)(i / ((size_t)C * HW));
    const size_t rem = i - (size_t)n * C * HW;
    const int c = (int)(rem >> 18);
    const int p = (int)(rem & (HW - 1));
    out[i] = means[((size_t)n * C + c) * 256 + (ids[(size_t)n * HW + p] & 255)];
}

// ---------------------------------------------------------------------------
extern "C" void kernel_launch(void* const* d_in, const int* in_sizes, int n_in,
                              void* d_out, int out_size, void* d_ws, size_t ws_size,
                              hipStream_t stream)
{
    (void)in_sizes; (void)n_in; (void)out_size; (void)ws_size;

    const float* x0  = (const float*)d_in[0];
    const int*   ids = (const int*)d_in[1];
    const float* W[10];
    const float* B[10];
    for (int i = 0; i < 10; ++i) {
        W[i] = (const float*)d_in[2 + 2 * i];
        B[i] = (const float*)d_in[3 + 2 * i];
    }

    const int N = 4;
    const size_t MAXACT = (size_t)N * 16 * 512 * 512;   // largest activation (elements)
    // ws layout: two f16 ping-pong activation buffers, one f32 conv-out buffer,
    // small means table. Total ~134 MB.
    _Float16* bufH0 = (_Float16*)d_ws;
    _Float16* bufH1 = bufH0 + MAXACT;
    float*    bufC  = (float*)(bufH1 + MAXACT);
    float*    means = bufC + MAXACT;                    // N*3*256 floats

    struct Layer { int Cin, Cout, Hin, Hout, k, tr, wshift; };
    const Layer L[10] = {
        {  3,  16, 512, 512, 7, 0, 9},   // 7x7 s1 reflect
        { 16,  32, 512, 256, 3, 0, 8},
        { 32,  64, 256, 128, 3, 0, 7},
        { 64, 128, 128,  64, 3, 0, 6},
        {128, 256,  64,  32, 3, 0, 5},
        {256, 128,  32,  64, 3, 1, 6},   // convT
        {128,  64,  64, 128, 3, 1, 7},
        { 64,  32, 128, 256, 3, 1, 8},
        { 32,  16, 256, 512, 3, 1, 9},
        { 16,   3, 512, 512, 7, 0, 9}    // 7x7 -> IN -> tanh
    };

    // Convert the network input to f16 (stored in bufH1; consumed by layer 0,
    // overwritten only after layer 1 has produced its own output).
    const int nin = N * 3 * 512 * 512;
    f32_to_f16<<<(nin + 255) / 256, 256, 0, stream>>>(x0, bufH1, nin);

    const _Float16* cur = bufH1;
    _Float16* bufs[2] = {bufH0, bufH1};
    int which = 0;

    for (int i = 0; i < 10; ++i) {
        const Layer& l = L[i];
        const int HWout = l.Hout * l.Hout;
        dim3 grid(HWout / 128, (l.Cout + 63) / 64, N);
        if (l.k == 7)
            conv_wmma<7, 1, 3, 0><<<grid, 256, 0, stream>>>(
                cur, W[i], B[i], bufC, l.Cin, l.Cout, l.Hin, l.Hin, l.Hout, l.Hout, l.wshift);
        else if (!l.tr)
            conv_wmma<3, 2, 1, 0><<<grid, 256, 0, stream>>>(
                cur, W[i], B[i], bufC, l.Cin, l.Cout, l.Hin, l.Hin, l.Hout, l.Hout, l.wshift);
        else
            conv_wmma<3, 2, 1, 1><<<grid, 256, 0, stream>>>(
                cur, W[i], B[i], bufC, l.Cin, l.Cout, l.Hin, l.Hin, l.Hout, l.Hout, l.wshift);

        if (i < 9) {
            inorm_act<<<N * l.Cout, 256, 0, stream>>>(bufC, nullptr, bufs[which], HWout, 0);
            cur = bufs[which];
            which ^= 1;
        } else {
            inorm_act<<<N * l.Cout, 256, 0, stream>>>(bufC, bufC, nullptr, HWout, 1);
        }
    }

    const int HW = 512 * 512;
    imean_stats<<<N * 3, 256, 0, stream>>>(bufC, ids, means);
    imean_scatter<<<(N * 3 * HW) / 256, 256, 0, stream>>>(means, ids, (float*)d_out);
}
